// PointAttentionNetwork_12446815224289
// MI455X (gfx1250) — matrix operand order repaired and checked by reading, not compiled, not run on hardware
//
#include <hip/hip_runtime.h>
#include <hip/hip_bf16.h>

typedef __attribute__((ext_vector_type(16))) __bf16 v16bf;
typedef __attribute__((ext_vector_type(8)))  __bf16 v8bf;
typedef __attribute__((ext_vector_type(8)))  float  v8f;
typedef __attribute__((ext_vector_type(4)))  float  v4f;

#define B_   4
#define C_   256
#define N_   4096
#define CR_  32
#define EPSB 1e-5f

static __device__ __forceinline__ v16bf cat8(v8bf a, v8bf b) {
    return __builtin_shufflevector(a, b, 0, 1, 2, 3, 4, 5, 6, 7,
                                         8, 9, 10, 11, 12, 13, 14, 15);
}

// ---------------------------------------------------------------- utilities

__global__ void cvt_bf16_kernel(const float* __restrict__ in, __bf16* __restrict__ out, int n) {
    int i = blockIdx.x * blockDim.x + threadIdx.x;
    if (i < n) out[i] = (__bf16)in[i];
}

// x [B,C,N] f32  ->  xT [B,N,C] bf16   (LDS tiled transpose)
__global__ void transpose_x_kernel(const float* __restrict__ x, __bf16* __restrict__ xT) {
    __shared__ float tile[32][33];
    int b  = blockIdx.z;
    int n0 = blockIdx.x * 32, c0 = blockIdx.y * 32;
    int tx = threadIdx.x, ty = threadIdx.y;                  // block (32,8)
    const float* xp = x + (size_t)b * C_ * N_;
    #pragma unroll
    for (int r = 0; r < 32; r += 8)
        tile[ty + r][tx] = xp[(size_t)(c0 + ty + r) * N_ + n0 + tx];
    __syncthreads();
    __bf16* op = xT + (size_t)b * N_ * C_;
    #pragma unroll
    for (int r = 0; r < 32; r += 8)
        op[(size_t)(n0 + ty + r) * C_ + c0 + tx] = (__bf16)tile[tx][ty + r];
}

// ------------------------------------------------ conv GEMM: Y[b,o,n] = W[o,:]·x[b,:,n]
// one wave -> one 16x16 (o,n) tile, K=C_=256 in 8 WMMA steps of 32
__global__ void __launch_bounds__(128) conv_gemm_kernel(
    const __bf16* __restrict__ Wb,       // [O, C] bf16
    const __bf16* __restrict__ xT,       // [B, N, C] bf16
    float* __restrict__ y,               // [B, O, N] f32
    int O) {
    int wave = (blockIdx.x * blockDim.x + threadIdx.x) >> 5;
    int lane = threadIdx.x & 31;
    int h = lane >> 4, l16 = lane & 15;

    int tiles_n = N_ >> 4;
    int per_b   = (O >> 4) * tiles_n;
    int b  = wave / per_b;
    int t  = wave % per_b;
    int ot = t / tiles_n, nt = t % tiles_n;

    const __bf16* wrow = Wb + (size_t)(ot * 16 + l16) * C_;          // A: row = o
    const __bf16* xrow = xT + ((size_t)b * N_ + nt * 16 + l16) * C_; // B: col = n

    v8f acc = {};
    #pragma unroll
    for (int kk = 0; kk < C_; kk += 32) {
        v8bf a0 = *(const v8bf*)(wrow + kk + h * 8);
        v8bf a1 = *(const v8bf*)(wrow + kk + h * 8 + 16);
        v16bf a  = cat8(a0, a1);
        v16bf bm = cat8(*(const v8bf*)(xrow + kk + h * 8), *(const v8bf*)(xrow + kk + h * 8 + 16));
        acc = __builtin_amdgcn_wmma_f32_16x16x32_bf16(false, a, false, bm,
                                                      (short)0, acc, false, false);
    }
    // D: VGPR r -> o = ot*16 + r + 8h ; lane -> n
    float* yp = y + ((size_t)b * O + ot * 16 + 8 * h) * N_ + nt * 16 + l16;
    #pragma unroll
    for (int r = 0; r < 8; ++r)
        yp[(size_t)r * N_] = acc[r];
}

// -------------------------------------------------- BN stats per out-channel
__global__ void bn_stats_kernel(const float* __restrict__ y, float* __restrict__ mean,
                                float* __restrict__ rstd, int O) {
    int o = blockIdx.x, tid = threadIdx.x;
    float s = 0.f, s2 = 0.f;
    for (int i = tid; i < B_ * N_; i += 256) {
        int b = i >> 12, n = i & (N_ - 1);
        float v = y[((size_t)b * O + o) * N_ + n];
        s += v; s2 += v * v;
    }
    __shared__ float sh0[256], sh1[256];
    sh0[tid] = s; sh1[tid] = s2; __syncthreads();
    for (int off = 128; off > 0; off >>= 1) {
        if (tid < off) { sh0[tid] += sh0[tid + off]; sh1[tid] += sh1[tid + off]; }
        __syncthreads();
    }
    if (tid == 0) {
        float m   = sh0[0] * (1.0f / (B_ * N_));
        float var = sh1[0] * (1.0f / (B_ * N_)) - m * m;
        mean[o] = m;
        rstd[o] = rsqrtf(var + EPSB);
    }
}

// BN+ReLU, write transposed [b,n,o] bf16  (for q and k^T, O = 32)
__global__ void bn_apply_qk_kernel(const float* __restrict__ y, const float* __restrict__ g,
                                   const float* __restrict__ bt, const float* __restrict__ mean,
                                   const float* __restrict__ rstd, __bf16* __restrict__ out) {
    int i = blockIdx.x * blockDim.x + threadIdx.x;   // B*N*32, o fastest
    int o = i & 31;
    int bn = i >> 5;
    int n = bn & (N_ - 1);
    int b = bn >> 12;
    float v = y[((size_t)b * CR_ + o) * N_ + n];
    float t = g[o] * ((v - mean[o]) * rstd[o]) + bt[o];
    out[i] = (__bf16)fmaxf(t, 0.f);
}

// BN+ReLU elementwise [b,c,n] bf16 (for d, O = 256)
__global__ void bn_apply_d_kernel(const float* __restrict__ y, const float* __restrict__ g,
                                  const float* __restrict__ bt, const float* __restrict__ mean,
                                  const float* __restrict__ rstd, __bf16* __restrict__ out) {
    int i = blockIdx.x * blockDim.x + threadIdx.x;   // B*C*N
    int c = (i >> 12) & (C_ - 1);
    float v = y[i];
    float t = g[c] * ((v - mean[c]) * rstd[c]) + bt[c];
    out[i] = (__bf16)fmaxf(t, 0.f);
}

// ------------------------------------------------ flash attention + residual
// per wave: 16 query rows, all 256 channels, online softmax, 64 keys per block
__global__ void __launch_bounds__(128) flash_attn_kernel(
    const __bf16* __restrict__ q,   // [B, N, 32]
    const __bf16* __restrict__ kt,  // [B, N, 32]   (K transposed)
    const __bf16* __restrict__ dv,  // [B, C, N]    (values, row-major in n)
    const float*  __restrict__ x,   // [B, C, N]
    float* __restrict__ out) {      // [B, C, N]
    int wave = threadIdx.x >> 5;
    int lane = threadIdx.x & 31;
    int h = lane >> 4, l16 = lane & 15;
    int tile = blockIdx.x * 4 + wave;
    int b  = tile >> 8;                 // N/16 = 256 row tiles per batch
    int i0 = (tile & 255) << 4;

    __shared__ __bf16 psh[4][16 * 64];  // wave-private P transpose scratch (2 KB/wave)
    __bf16* ps = psh[wave];

    // Q as A-operand: lane row = i0 + l16, K(e) = 8h + {0..7, 16..23}
    const __bf16* qrow = q + ((size_t)b * N_ + i0 + l16) * CR_;
    v16bf qa = cat8(*(const v8bf*)(qrow + h * 8),
                    *(const v8bf*)(qrow + h * 8 + 16));

    float m[8], l[8];
    #pragma unroll
    for (int r = 0; r < 8; ++r) { m[r] = -3.0e38f; l[r] = 0.f; }
    v8f acc[16];
    #pragma unroll
    for (int t = 0; t < 16; ++t) acc[t] = (v8f){0, 0, 0, 0, 0, 0, 0, 0};

    for (int j0 = 0; j0 < N_; j0 += 64) {
        // logits for 4 consecutive 16-key tiles
        const __bf16* k0 = kt + ((size_t)b * N_ + j0 + l16) * CR_ + 8 * h;
        v8f s[4];
        v8f z = {};
        #pragma unroll
        for (int u = 0; u < 4; ++u) {
            v16bf kb = cat8(*(const v8bf*)(k0 + (size_t)u * 16 * CR_), *(const v8bf*)(k0 + (size_t)u * 16 * CR_ + 16));
            s[u] = __builtin_amdgcn_wmma_f32_16x16x32_bf16(false, qa, false, kb,
                                                           (short)0, z, false, false);
        }

        // row max across 64 keys (rows per-VGPR, broadcast within 16-lane half)
        float mx[8];
        #pragma unroll
        for (int r = 0; r < 8; ++r)
            mx[r] = fmaxf(fmaxf(s[0][r], s[1][r]), fmaxf(s[2][r], s[3][r]));
        #pragma unroll
        for (int off = 1; off < 16; off <<= 1) {
            #pragma unroll
            for (int r = 0; r < 8; ++r) mx[r] = fmaxf(mx[r], __shfl_xor(mx[r], off, 32));
        }

        float scale[8], rs[8];
        float p[4][8];
        #pragma unroll
        for (int r = 0; r < 8; ++r) {
            float mn = fmaxf(m[r], mx[r]);
            scale[r] = __expf(m[r] - mn);
            m[r] = mn;
            float acc_rs = 0.f;
            #pragma unroll
            for (int u = 0; u < 4; ++u) {
                p[u][r] = __expf(s[u][r] - mn);
                acc_rs += p[u][r];
            }
            rs[r] = acc_rs;
        }
        #pragma unroll
        for (int off = 1; off < 16; off <<= 1) {
            #pragma unroll
            for (int r = 0; r < 8; ++r) rs[r] += __shfl_xor(rs[r], off, 32);
        }
        #pragma unroll
        for (int r = 0; r < 8; ++r) l[r] = l[r] * scale[r] + rs[r];

        // transpose P (D-layout) -> A-layout via wave-private LDS, [16 rows][64 cols] bf16
        #pragma unroll
        for (int u = 0; u < 4; ++u) {
            #pragma unroll
            for (int r = 0; r < 8; ++r)
                ps[(r + 8 * h) * 64 + u * 16 + l16] = (__bf16)p[u][r];
        }
        asm volatile("s_wait_dscnt 0" ::: "memory");
        const __bf16* prow = ps + l16 * 64 + h * 8;
        v16bf pa0 = cat8(*(const v8bf*)(prow),      *(const v8bf*)(prow + 16));
        v16bf pa1 = cat8(*(const v8bf*)(prow + 32), *(const v8bf*)(prow + 48));

        // rescale accumulators and run PV WMMAs over the 256 channels
        const __bf16* vb = dv + ((size_t)b * C_ + l16) * N_ + j0 + 8 * h;
        #pragma unroll
        for (int t = 0; t < 16; ++t) {
            #pragma unroll
            for (int r = 0; r < 8; ++r) acc[t][r] *= scale[r];
            v16bf vt0 = cat8(*(const v8bf*)(vb + (size_t)t * 16 * N_), *(const v8bf*)(vb + (size_t)t * 16 * N_ + 16));
            v16bf vt1 = cat8(*(const v8bf*)(vb + (size_t)t * 16 * N_ + 32), *(const v8bf*)(vb + (size_t)t * 16 * N_ + 48));
            acc[t] = __builtin_amdgcn_wmma_f32_16x16x32_bf16(false, pa0, false, vt0,
                                                             (short)0, acc[t], false, false);
            acc[t] = __builtin_amdgcn_wmma_f32_16x16x32_bf16(false, pa1, false, vt1,
                                                             (short)0, acc[t], false, false);
        }
    }

    float rcl[8];
    #pragma unroll
    for (int r = 0; r < 8; ++r) rcl[r] = 1.0f / l[r];

    int ib = i0 + 8 * h;                 // rows r=0..7 contiguous in memory
    #pragma unroll
    for (int t = 0; t < 16; ++t) {
        int c = t * 16 + l16;
        size_t idx = ((size_t)b * C_ + c) * N_ + ib;
        v4f x0 = *(const v4f*)(x + idx);
        v4f x1 = *(const v4f*)(x + idx + 4);
        v4f o0, o1;
        #pragma unroll
        for (int r = 0; r < 4; ++r) {
            o0[r] = x0[r] + acc[t][r] * rcl[r];
            o1[r] = x1[r] + acc[t][r + 4] * rcl[r + 4];
        }
        *(v4f*)(out + idx)     = o0;
        *(v4f*)(out + idx + 4) = o1;
    }
}

// ---------------------------------------------------------------- launcher

extern "C" void kernel_launch(void* const* d_in, const int* in_sizes, int n_in,
                              void* d_out, int out_size, void* d_ws, size_t ws_size,
                              hipStream_t stream) {
    const float* x  = (const float*)d_in[0];
    const float* w1 = (const float*)d_in[1];
    const float* w2 = (const float*)d_in[2];
    const float* w3 = (const float*)d_in[3];
    const float* g1 = (const float*)d_in[4];
    const float* b1 = (const float*)d_in[5];
    const float* g2 = (const float*)d_in[6];
    const float* b2 = (const float*)d_in[7];
    const float* g3 = (const float*)d_in[8];
    const float* b3 = (const float*)d_in[9];
    float* out = (float*)d_out;

    char* wp = (char*)d_ws;
    auto alloc = [&](size_t bytes) -> char* {
        char* p = wp; wp += (bytes + 255) & ~(size_t)255; return p;
    };
    float*  y1    = (float*)  alloc(sizeof(float) * B_ * CR_ * N_);
    float*  y2    = (float*)  alloc(sizeof(float) * B_ * CR_ * N_);
    float*  y3    = (float*)  alloc(sizeof(float) * B_ * C_  * N_);
    __bf16* xT    = (__bf16*) alloc(sizeof(__bf16) * B_ * N_ * C_);
    __bf16* w1b   = (__bf16*) alloc(sizeof(__bf16) * CR_ * C_);
    __bf16* w2b   = (__bf16*) alloc(sizeof(__bf16) * CR_ * C_);
    __bf16* w3b   = (__bf16*) alloc(sizeof(__bf16) * C_ * C_);
    __bf16* qb    = (__bf16*) alloc(sizeof(__bf16) * B_ * N_ * CR_);
    __bf16* ktb   = (__bf16*) alloc(sizeof(__bf16) * B_ * N_ * CR_);
    __bf16* db    = (__bf16*) alloc(sizeof(__bf16) * B_ * C_ * N_);
    float*  mean1 = (float*)  alloc(sizeof(float) * CR_);
    float*  rstd1 = (float*)  alloc(sizeof(float) * CR_);
    float*  mean2 = (float*)  alloc(sizeof(float) * CR_);
    float*  rstd2 = (float*)  alloc(sizeof(float) * CR_);
    float*  mean3 = (float*)  alloc(sizeof(float) * C_);
    float*  rstd3 = (float*)  alloc(sizeof(float) * C_);

    // weights -> bf16
    cvt_bf16_kernel<<<(CR_ * C_ + 255) / 256, 256, 0, stream>>>(w1, w1b, CR_ * C_);
    cvt_bf16_kernel<<<(CR_ * C_ + 255) / 256, 256, 0, stream>>>(w2, w2b, CR_ * C_);
    cvt_bf16_kernel<<<(C_ * C_ + 255) / 256, 256, 0, stream>>>(w3, w3b, C_ * C_);

    // x -> xT bf16
    transpose_x_kernel<<<dim3(N_ / 32, C_ / 32, B_), dim3(32, 8), 0, stream>>>(x, xT);

    // three conv GEMMs (WMMA)
    conv_gemm_kernel<<<(B_ * (CR_ / 16) * (N_ / 16)) / 4, 128, 0, stream>>>(w1b, xT, y1, CR_);
    conv_gemm_kernel<<<(B_ * (CR_ / 16) * (N_ / 16)) / 4, 128, 0, stream>>>(w2b, xT, y2, CR_);
    conv_gemm_kernel<<<(B_ * (C_  / 16) * (N_ / 16)) / 4, 128, 0, stream>>>(w3b, xT, y3, C_);

    // batchnorm stats + apply
    bn_stats_kernel<<<CR_, 256, 0, stream>>>(y1, mean1, rstd1, CR_);
    bn_stats_kernel<<<CR_, 256, 0, stream>>>(y2, mean2, rstd2, CR_);
    bn_stats_kernel<<<C_,  256, 0, stream>>>(y3, mean3, rstd3, C_);

    bn_apply_qk_kernel<<<(B_ * N_ * CR_) / 256, 256, 0, stream>>>(y1, g1, b1, mean1, rstd1, qb);
    bn_apply_qk_kernel<<<(B_ * N_ * CR_) / 256, 256, 0, stream>>>(y2, g2, b2, mean2, rstd2, ktb);
    bn_apply_d_kernel<<<(B_ * C_ * N_) / 256, 256, 0, stream>>>(y3, g3, b3, mean3, rstd3, db);

    // flash attention + residual (WMMA)
    flash_attn_kernel<<<(B_ * (N_ / 16)) / 4, 128, 0, stream>>>(qb, ktb, db, x, out);
}
